// SeqProbMVSNet_82721070121377
// MI455X (gfx1250) — compile-verified, matrix operand order checked by reading, not run on hardware
//
#include <hip/hip_runtime.h>
#include <hip/hip_bf16.h>
#include <math.h>

// Problem dims (compile-time constants from the reference)
#define Bsz 1
#define Cch 32
#define Dd  48
#define Hh  128
#define Ww  160
#define KTOT 864          // Cch * 27
#define KCH  27           // number of 32-deep K chunks; chunk kc == spatial tap kc
#define WT   32           // conv1 w-tile width per block
#define NWT  (Ww / WT)    // 5

typedef __attribute__((ext_vector_type(16))) _Float16 v16h;
typedef __attribute__((ext_vector_type(8)))  float    v8f;

// K ordering (0..863):  K = tap*32 + ic,  tap = kd*9 + kh*3 + kw  (channel-minor)
// This makes each lane's WMMA A-fragment two contiguous 8-channel runs.

// ---------------- workspace layout (bytes) ----------------
// mats      : 2 views * 12 floats                        @ 0
// wfrag     : 27 * 32 * 16 halves = 27648 B              @ 256
// var (f16) : [d][h][w][ic]  32*48*128*160*2 B           @ 32768
// cost1(f16): [d][h][w][oc]  8*48*128*160*2 B
// cost2(f32): [d][h][w]      48*128*160*4 B
#define WS_MAT_OFF    ((size_t)0)
#define WS_WFRAG_OFF  ((size_t)256)
#define WS_VAR_OFF    ((size_t)32768)
#define WS_COST1_OFF  (WS_VAR_OFF + (size_t)Cch*Dd*Hh*Ww*2)
#define WS_COST2_OFF  (WS_COST1_OFF + (size_t)8*Dd*Hh*Ww*2)

// ---------------- small device helpers ----------------
__device__ inline void combine_proj(const float* p, float* out) {
    // p: [2][4][4]; p[0]=extrinsic E, p[1]=intrinsic K
    const float* E  = p;
    const float* Km = p + 16;
    #pragma unroll
    for (int i = 0; i < 16; i++) out[i] = E[i];
    #pragma unroll
    for (int i = 0; i < 3; i++)
        #pragma unroll
        for (int j = 0; j < 4; j++) {
            float s = 0.f;
            #pragma unroll
            for (int k = 0; k < 3; k++) s += Km[i*4+k] * E[k*4+j];
            out[i*4+j] = s;
        }
}

__device__ inline void invert4(const float* a_in, float* inv) {
    float a[16];
    #pragma unroll
    for (int i = 0; i < 16; i++) { a[i] = a_in[i]; inv[i] = (i % 5 == 0) ? 1.f : 0.f; }
    for (int col = 0; col < 4; col++) {
        int piv = col;
        float best = fabsf(a[col*4+col]);
        for (int r = col+1; r < 4; r++) {
            float v = fabsf(a[r*4+col]);
            if (v > best) { best = v; piv = r; }
        }
        if (piv != col) {
            for (int j = 0; j < 4; j++) {
                float t = a[col*4+j]; a[col*4+j] = a[piv*4+j]; a[piv*4+j] = t;
                t = inv[col*4+j]; inv[col*4+j] = inv[piv*4+j]; inv[piv*4+j] = t;
            }
        }
        float rd = 1.f / a[col*4+col];
        for (int j = 0; j < 4; j++) { a[col*4+j] *= rd; inv[col*4+j] *= rd; }
        for (int r = 0; r < 4; r++) if (r != col) {
            float f = a[r*4+col];
            for (int j = 0; j < 4; j++) {
                a[r*4+j]  -= f * a[col*4+j];
                inv[r*4+j] -= f * inv[col*4+j];
            }
        }
    }
}

__device__ inline void matmul4(const float* A, const float* B, float* C) {
    #pragma unroll
    for (int i = 0; i < 4; i++)
        #pragma unroll
        for (int j = 0; j < 4; j++) {
            float s = 0.f;
            #pragma unroll
            for (int k = 0; k < 4; k++) s += A[i*4+k] * B[k*4+j];
            C[i*4+j] = s;
        }
}

// ---------------- kernel 1: projection matrices + weight fragment swizzle ----------------
__global__ void prep_kernel(const float* __restrict__ ref_proj,
                            const float* __restrict__ src_proj1,
                            const float* __restrict__ src_proj2,
                            const float* __restrict__ w1,
                            char* __restrict__ ws) {
    float*    mats  = (float*)(ws + WS_MAT_OFF);
    _Float16* wfrag = (_Float16*)(ws + WS_WFRAG_OFF);

    if (threadIdx.x == 0) {
        float rp[16], inv[16];
        combine_proj(ref_proj, rp);
        invert4(rp, inv);
        const float* sps[2] = { src_proj1, src_proj2 };
        for (int v = 0; v < 2; v++) {
            float sp[16], m[16];
            combine_proj(sps[v], sp);
            matmul4(sp, inv, m);
            float* o = mats + v * 12;
            for (int i = 0; i < 3; i++)
                for (int j = 0; j < 3; j++) o[i*3+j] = m[i*4+j];
            o[9] = m[3]; o[10] = m[7]; o[11] = m[11];
        }
    }

    // B-fragment layout: wfrag[(kc*32 + lane)*16 + e]
    // 16-bit matrix element map: e = 2*j + t (j = VGPR, t = low/high half):
    //   klocal(=ic) = (j>>2)*16 + ((lane&16)?8:0) + 2*(j&3) + t ;  N (oc col) = lane&15
    for (int idx = threadIdx.x; idx < KCH * 32 * 16; idx += blockDim.x) {
        int e    = idx & 15;
        int lane = (idx >> 4) & 31;
        int kc   = idx >> 9;                  // tap = kd*9 + kh*3 + kw
        int j = e >> 1, t = e & 1;
        int ic = ((j >> 2) * 16) + ((lane & 16) ? 8 : 0) + 2 * (j & 3) + t;
        int n  = lane & 15;
        float val = 0.f;
        if (n < 8) val = w1[(n * Cch + ic) * 27 + kc];   // w1: (8,32,3,3,3)
        wfrag[idx] = (_Float16)val;
    }
}

// ---------------- kernel 2: warp + 3-view variance (f16, channel-last) ----------------
__global__ void variance_kernel(const float* __restrict__ ref,
                                const float* __restrict__ src1,
                                const float* __restrict__ src2,
                                const float* __restrict__ depth,
                                char* __restrict__ ws) {
    const float* mats = (const float*)(ws + WS_MAT_OFF);
    uint4*       varq = (uint4*)(ws + WS_VAR_OFF);      // [d][h][w][ic] -> 4 uint4 per voxel

    int idx = blockIdx.x * blockDim.x + threadIdx.x;
    if (idx >= Dd * Hh * Ww) return;
    int w = idx % Ww;
    int h = (idx / Ww) % Hh;

    float dep = depth[idx];
    float x = (float)w, y = (float)h;

    int   xi0[2], xi1[2], yi0[2], yi1[2];
    float m00[2], m01[2], m10[2], m11[2], fwx[2], fwy[2];
    #pragma unroll
    for (int v = 0; v < 2; v++) {
        const float* mm = mats + v * 12;
        float X = mm[0]*x + mm[1]*y + mm[2];
        float Y = mm[3]*x + mm[4]*y + mm[5];
        float Z = mm[6]*x + mm[7]*y + mm[8];
        X = X * dep + mm[9];
        Y = Y * dep + mm[10];
        Z = Z * dep + mm[11];
        if (fabsf(Z) < 1e-6f) Z = 1e-6f;
        float pxv = X / Z, pyv = Y / Z;
        float x0 = floorf(pxv), y0 = floorf(pyv);
        fwx[v] = pxv - x0; fwy[v] = pyv - y0;
        float x1 = x0 + 1.f, y1 = y0 + 1.f;
        m00[v] = (x0 >= 0.f && x0 <= (float)(Ww-1) && y0 >= 0.f && y0 <= (float)(Hh-1)) ? 1.f : 0.f;
        m01[v] = (x1 >= 0.f && x1 <= (float)(Ww-1) && y0 >= 0.f && y0 <= (float)(Hh-1)) ? 1.f : 0.f;
        m10[v] = (x0 >= 0.f && x0 <= (float)(Ww-1) && y1 >= 0.f && y1 <= (float)(Hh-1)) ? 1.f : 0.f;
        m11[v] = (x1 >= 0.f && x1 <= (float)(Ww-1) && y1 >= 0.f && y1 <= (float)(Hh-1)) ? 1.f : 0.f;
        xi0[v] = (int)fminf(fmaxf(x0, 0.f), (float)(Ww-1));
        xi1[v] = (int)fminf(fmaxf(x1, 0.f), (float)(Ww-1));
        yi0[v] = (int)fminf(fmaxf(y0, 0.f), (float)(Hh-1));
        yi1[v] = (int)fminf(fmaxf(y1, 0.f), (float)(Hh-1));
    }

    const float* srcs[2] = { src1, src2 };
    const float inv3 = 1.f / 3.f;

    // process channels in groups of 8 -> one b128 store per group
    for (int g = 0; g < 4; g++) {
        union { uint4 q; _Float16 h[8]; } o;
        #pragma unroll
        for (int cl = 0; cl < 8; cl++) {
            int c = g * 8 + cl;
            float r  = ref[c * Hh * Ww + h * Ww + w];
            float s  = r;
            float sq = r * r;
            #pragma unroll
            for (int v = 0; v < 2; v++) {
                const float* im = srcs[v] + c * Hh * Ww;
                float v00 = im[yi0[v]*Ww + xi0[v]] * m00[v];
                float v01 = im[yi0[v]*Ww + xi1[v]] * m01[v];
                float v10 = im[yi1[v]*Ww + xi0[v]] * m10[v];
                float v11 = im[yi1[v]*Ww + xi1[v]] * m11[v];
                float wxv = fwx[v], wyv = fwy[v];
                float val = v00 * (1.f-wxv) * (1.f-wyv) + v01 * wxv * (1.f-wyv)
                          + v10 * (1.f-wxv) * wyv       + v11 * wxv * wyv;
                s += val; sq += val * val;
            }
            float mean = s * inv3;
            o.h[cl] = (_Float16)(sq * inv3 - mean * mean);
        }
        varq[(size_t)idx * 4 + g] = o.q;
    }
}

// ---------------- kernel 3: conv1 (32->8, 3x3x3) as implicit GEMM via WMMA ----------------
// Block = 64 threads (2 waves). Each block: fixed (d,h), 32-wide w-tile.
// Halo in LDS: [dd][hh][ww][ic] channel-last, 9*34*64 B = 19584 B.
// Halo staged with GLOBAL_LOAD_ASYNC_TO_LDS_B128 (ASYNCcnt path); zero-pad
// positions are written with a regular LDS store by the EXEC-inverse lanes.
// A fragment per WMMA = two ds_load_b128 (contiguous 8-channel runs).
__global__ void __launch_bounds__(64)
conv1_wmma_kernel(const float* __restrict__ b1, char* __restrict__ ws) {
    __shared__ uint4 halo4[9 * (WT + 2) * 4];

    const uint4* varq  = (const uint4*)(ws + WS_VAR_OFF);
    const v16h*  wfrag = (const v16h*)(ws + WS_WFRAG_OFF);
    _Float16*    cost1 = (_Float16*)(ws + WS_COST1_OFF);

    int block = blockIdx.x;            // (d*Hh + h)*NWT + wt
    int wt = block % NWT;
    int h  = (block / NWT) % Hh;
    int d  = block / (NWT * Hh);
    int wbase = wt * WT;
    int tid = threadIdx.x;

    __builtin_prefetch((const void*)(wfrag + (tid & 31)), 0, 0);

    // LDS byte address of halo4 (aperture keeps the LDS offset in addr[31:0])
    unsigned lds_base = (unsigned)(size_t)(&halo4[0]);

    // Cooperative halo stage: var[d-1..d+1][h-1..h+1][wbase-1..wbase+WT][0..31]
    for (int i = tid; i < 9 * (WT + 2) * 4; i += 64) {
        int quad = i & 3;              // which 8-channel group
        int pos  = i >> 2;             // (dd*3+hh)*(WT+2) + ww
        int ww = pos % (WT + 2);
        int r  = pos / (WT + 2);
        int hh = r % 3;
        int dd = r / 3;
        int gw = wbase + ww - 1;
        int gh = h + hh - 1;
        int gd = d + dd - 1;
        unsigned lds_off = lds_base + (unsigned)i * 16u;
        if ((unsigned)gw < (unsigned)Ww && (unsigned)gh < (unsigned)Hh &&
            (unsigned)gd < (unsigned)Dd) {
            const uint4* gp = varq + (size_t)((gd * Hh + gh) * Ww + gw) * 4 + quad;
            // async DMA: global -> LDS, no VGPR staging, tracked by ASYNCcnt
            asm volatile("global_load_async_to_lds_b128 %0, %1, off"
                         :
                         : "v"(lds_off), "v"((unsigned long long)(size_t)gp)
                         : "memory");
        } else {
            halo4[i] = make_uint4(0u, 0u, 0u, 0u);   // zero pad (ds_store_b128)
        }
    }
    // ASYNCcnt is not covered by the barrier's dscnt wait: drain it explicitly.
    asm volatile("s_wait_asynccnt 0x0" ::: "memory");
    __syncthreads();

    int lane  = tid & 31;
    int wave  = tid >> 5;                  // 0..1
    int lw    = wave * 16 + (lane & 15);   // local voxel (A/D row M base) 0..31
    int half8 = (lane & 16) ? 8 : 0;       // upper-half lanes take ic+8 / row M+8

    v8f acc = {};
    #pragma unroll
    for (int kc = 0; kc < KCH; kc++) {
        int kd  = kc / 9;
        int kh2 = (kc / 3) % 3;
        int kw  = kc % 3;
        int rowq = ((kd * 3 + kh2) * (WT + 2) + lw + kw) * 4;   // uint4 index of voxel row
        union { uint4 q[2]; v16h v; } af;
        af.q[0] = halo4[rowq + (half8 >> 3)];       // ic in [half8, half8+8)
        af.q[1] = halo4[rowq + 2 + (half8 >> 3)];   // ic in [16+half8, 16+half8+8)
        v16h bf = wfrag[kc * 32 + lane];
        acc = __builtin_amdgcn_wmma_f32_16x16x32_f16(
                  /*neg_a=*/false, af.v, /*neg_b=*/false, bf,
                  /*c_mod=*/(short)0, acc, /*reuse_a=*/false, /*reuse_b=*/false);
    }

    // D layout: VGPR j -> M = j + ((lane&16)?8:0), N = lane&15
    int oc = lane & 15;
    if (oc < 8) {
        float bias = b1[oc];
        #pragma unroll
        for (int j = 0; j < 8; j++) {
            int M  = j + half8;
            int gw = wbase + wave * 16 + M;
            float v = acc[j] + bias;
            v = fmaxf(v, 0.f);   // ReLU
            cost1[(size_t)((d * Hh + h) * Ww + gw) * 8 + oc] = (_Float16)v;
        }
    }
}

// ---------------- kernel 4: conv2 (8->1, 3x3x3), channel-last cost1 ----------------
__global__ void conv2_kernel(const float* __restrict__ w2,
                             const float* __restrict__ b2,
                             char* __restrict__ ws) {
    const uint4* cost1q = (const uint4*)(ws + WS_COST1_OFF);   // 8 halves per voxel
    float*       cost2  = (float*)(ws + WS_COST2_OFF);

    int idx = blockIdx.x * blockDim.x + threadIdx.x;
    if (idx >= Dd * Hh * Ww) return;
    int w = idx % Ww;
    int h = (idx / Ww) % Hh;
    int d = idx / (Ww * Hh);

    float s = b2[0];
    #pragma unroll
    for (int kd = 0; kd < 3; kd++) {
        int gd = d + kd - 1;
        if ((unsigned)gd >= (unsigned)Dd) continue;
        #pragma unroll
        for (int kh = 0; kh < 3; kh++) {
            int gh = h + kh - 1;
            if ((unsigned)gh >= (unsigned)Hh) continue;
            #pragma unroll
            for (int kw = 0; kw < 3; kw++) {
                int gw = w + kw - 1;
                if ((unsigned)gw >= (unsigned)Ww) continue;
                union { uint4 q; _Float16 hh[8]; } c;
                c.q = cost1q[(size_t)((gd * Hh + gh) * Ww + gw)];
                int tap = kd * 9 + kh * 3 + kw;
                #pragma unroll
                for (int oc = 0; oc < 8; oc++)
                    s += (float)c.hh[oc] * w2[oc * 27 + tap];   // w2: (1,8,3,3,3)
            }
        }
    }
    cost2[idx] = s;
}

// ---------------- kernel 5: softmax over depth ----------------
__global__ void softmax_kernel(char* __restrict__ ws, float* __restrict__ out) {
    const float* cost2 = (const float*)(ws + WS_COST2_OFF);
    int idx = blockIdx.x * blockDim.x + threadIdx.x;
    if (idx >= Hh * Ww) return;

    float mx = -INFINITY;
    for (int d = 0; d < Dd; d++) mx = fmaxf(mx, cost2[d * Hh * Ww + idx]);
    float s = 0.f;
    for (int d = 0; d < Dd; d++) s += __expf(cost2[d * Hh * Ww + idx] - mx);
    float inv = 1.f / s;
    for (int d = 0; d < Dd; d++)
        out[d * Hh * Ww + idx] = __expf(cost2[d * Hh * Ww + idx] - mx) * inv;
}

// ---------------- host launch ----------------
extern "C" void kernel_launch(void* const* d_in, const int* in_sizes, int n_in,
                              void* d_out, int out_size, void* d_ws, size_t ws_size,
                              hipStream_t stream) {
    const float* ref_fea  = (const float*)d_in[0];
    const float* src_fea1 = (const float*)d_in[1];
    const float* src_fea2 = (const float*)d_in[2];
    const float* ref_proj = (const float*)d_in[3];
    const float* sp1      = (const float*)d_in[4];
    const float* sp2      = (const float*)d_in[5];
    const float* depth    = (const float*)d_in[6];
    const float* w1       = (const float*)d_in[7];
    const float* b1       = (const float*)d_in[8];
    const float* w2       = (const float*)d_in[9];
    const float* b2       = (const float*)d_in[10];
    char*  ws  = (char*)d_ws;
    float* out = (float*)d_out;

    prep_kernel<<<1, 128, 0, stream>>>(ref_proj, sp1, sp2, w1, ws);

    int nvox = Dd * Hh * Ww;
    variance_kernel<<<(nvox + 255) / 256, 256, 0, stream>>>(ref_fea, src_fea1, src_fea2,
                                                            depth, ws);

    conv1_wmma_kernel<<<Dd * Hh * NWT, 64, 0, stream>>>(b1, ws);

    conv2_kernel<<<(nvox + 255) / 256, 256, 0, stream>>>(w2, b2, ws);

    softmax_kernel<<<(Hh * Ww + 255) / 256, 256, 0, stream>>>(ws, out);
}
